// MoE_ddG_NET_17935783428601
// MI455X (gfx1250) — matrix operand order repaired and checked by reading, not compiled, not run on hardware
//
#include <hip/hip_runtime.h>
#include <stdint.h>

#define N_TOK 16384
#define DMODEL 1024
#define NEXP 8
#define TOPK 2
#define FDIM 128
#define NOISE_EPS 0.01f
#define SCALE 0.5f
#define BTOK 64                         // tokens per expert block (4 M-tiles)

typedef __attribute__((ext_vector_type(16))) __bf16 v16bf;
typedef __attribute__((ext_vector_type(8)))  __bf16 v8bf;
typedef __attribute__((ext_vector_type(4)))  __bf16 v4bf;
typedef __attribute__((ext_vector_type(8)))  float  v8f;

#define SHUF16(lo, hi) __builtin_shufflevector((lo), (hi), 0,1,2,3,4,5,6,7,8,9,10,11,12,13,14,15)

// ---------------- prep kernels ----------------

__global__ __launch_bounds__(256) void k_zero_counts(int* counts) {
  if (threadIdx.x < NEXP) counts[threadIdx.x] = 0;
}

// fused: out = x (residual base) and xb = bf16(x), single pass over x
__global__ __launch_bounds__(256) void k_prep_x(const float* __restrict__ x,
                                                __bf16* __restrict__ xb,
                                                float* __restrict__ out) {
  size_t i = (size_t)blockIdx.x * blockDim.x + threadIdx.x;   // N*D/4 threads
  float4 v = ((const float4*)x)[i];
  ((float4*)out)[i] = v;
  v4bf o = { (__bf16)v.x, (__bf16)v.y, (__bf16)v.z, (__bf16)v.w };
  ((v4bf*)xb)[i] = o;
}

// W_down [E,D,F] f32 -> wd_t [E,F,D] bf16 (K-contiguous for WMMA B operand)
__global__ __launch_bounds__(256) void k_prep_wd(const float* __restrict__ Wd,
                                                 __bf16* __restrict__ wd_t) {
  int i = blockIdx.x * blockDim.x + threadIdx.x;              // E*F*D threads
  int d = i % DMODEL; int r = i / DMODEL;
  int f = r % FDIM;   int e = r / FDIM;
  wd_t[i] = (__bf16)Wd[((size_t)e * DMODEL + d) * FDIM + f];
}

// W_up [E,F,D] f32 -> wu_t [E,D,F] bf16
__global__ __launch_bounds__(256) void k_prep_wu(const float* __restrict__ Wu,
                                                 __bf16* __restrict__ wu_t) {
  int i = blockIdx.x * blockDim.x + threadIdx.x;              // E*D*F threads
  int f = i % FDIM;  int r = i / FDIM;
  int d = r % DMODEL; int e = r / DMODEL;
  wu_t[i] = (__bf16)Wu[((size_t)e * FDIM + f) * DMODEL + d];
}

// ---------------- gating: one wave32 per token ----------------

__global__ __launch_bounds__(256) void k_gate(const float* __restrict__ x,
                                              const float* __restrict__ noise,
                                              const float* __restrict__ router,
                                              const float* __restrict__ w_noise,
                                              int* __restrict__ counts,
                                              int* __restrict__ tok_ids,
                                              float* __restrict__ gate_v) {
  int tok  = (blockIdx.x * blockDim.x + threadIdx.x) >> 5;
  int lane = threadIdx.x & 31;
  const float* xr = x + (size_t)tok * DMODEL;

  float accC[NEXP]; float accS[NEXP];
#pragma unroll
  for (int e = 0; e < NEXP; ++e) { accC[e] = 0.f; accS[e] = 0.f; }

#pragma unroll 2
  for (int it = 0; it < DMODEL / 128; ++it) {
    int d0 = it * 128 + lane * 4;
    float4 xv = *(const float4*)(xr + d0);
    const float* xs = (const float*)&xv;
#pragma unroll
    for (int j = 0; j < 4; ++j) {
      float r8[NEXP], s8[NEXP];
      *(float4*)&r8[0] = *(const float4*)(router  + (size_t)(d0 + j) * NEXP);
      *(float4*)&r8[4] = *(const float4*)(router  + (size_t)(d0 + j) * NEXP + 4);
      *(float4*)&s8[0] = *(const float4*)(w_noise + (size_t)(d0 + j) * NEXP);
      *(float4*)&s8[4] = *(const float4*)(w_noise + (size_t)(d0 + j) * NEXP + 4);
#pragma unroll
      for (int e = 0; e < NEXP; ++e) {
        accC[e] = fmaf(xs[j], r8[e], accC[e]);
        accS[e] = fmaf(xs[j], s8[e], accS[e]);
      }
    }
  }
  // butterfly reduce across the wave (all lanes end with full sums)
#pragma unroll
  for (int off = 16; off > 0; off >>= 1) {
#pragma unroll
    for (int e = 0; e < NEXP; ++e) {
      accC[e] += __shfl_xor(accC[e], off, 32);
      accS[e] += __shfl_xor(accS[e], off, 32);
    }
  }

  if (lane == 0) {
    float nl[NEXP];
#pragma unroll
    for (int e = 0; e < NEXP; ++e) {
      float s  = accS[e];
      float sp = (s > 20.f) ? s : log1pf(expf(s));           // softplus
      nl[e] = accC[e] + noise[(size_t)tok * NEXP + e] * (sp + NOISE_EPS);
    }
    // top-2 (first occurrence wins ties, matching lax.top_k)
    int i0 = 0; float v0 = nl[0];
#pragma unroll
    for (int e = 1; e < NEXP; ++e) if (nl[e] > v0) { v0 = nl[e]; i0 = e; }
    int i1 = -1; float v1 = -3.0e38f;
#pragma unroll
    for (int e = 0; e < NEXP; ++e) if (e != i0 && nl[e] > v1) { v1 = nl[e]; i1 = e; }
    // softmax over the two selected logits
    float e1 = expf(v1 - v0);
    float g0 = 1.f / (1.f + e1);
    float g1 = e1 * g0;

    int p0 = atomicAdd(&counts[i0], 1);
    tok_ids[i0 * N_TOK + p0] = tok;  gate_v[i0 * N_TOK + p0] = g0;
    int p1 = atomicAdd(&counts[i1], 1);
    tok_ids[i1 * N_TOK + p1] = tok;  gate_v[i1 * N_TOK + p1] = g1;
  }
}

// -------- expert FFN: 64-token tile per block, 8 waves, 2x2 wave tiles --------

__global__ __launch_bounds__(256) void k_expert(const __bf16* __restrict__ xb,
                                                const __bf16* __restrict__ wd,   // [E,F,D]
                                                const __bf16* __restrict__ wu,   // [E,D,F]
                                                const float* __restrict__ b_down,// [E,F]
                                                const float* __restrict__ b_up,  // [E,D]
                                                const int*   __restrict__ counts,
                                                const int*   __restrict__ tok_ids,
                                                const float* __restrict__ gate_v,
                                                float* __restrict__ out) {
  int e     = blockIdx.x >> 8;         // / (N_TOK/BTOK)
  int tile  = blockIdx.x & 255;
  int cnt   = counts[e];
  int tbase = tile * BTOK;
  if (tbase >= cnt) return;            // uniform per block -> EXEC stays full

  __shared__ int   tok_s[BTOK];
  __shared__ float g_s[BTOK];
  __shared__ __bf16 xs[BTOK * DMODEL] __attribute__((aligned(32)));  // 128 KB x tile
  __shared__ __bf16 hsh[BTOK * FDIM]  __attribute__((aligned(32)));  // 16 KB hidden tile

  int tid = threadIdx.x;
  if (tid < BTOK) {
    int slot = tbase + tid;
    if (slot < cnt) {
      tok_s[tid] = tok_ids[e * N_TOK + slot];
      g_s[tid]   = gate_v[e * N_TOK + slot];
    } else {                           // pad row: valid address, zero gate
      tok_s[tid] = tok_ids[e * N_TOK + tbase];
      g_s[tid]   = 0.f;
    }
  }
  __syncthreads();

  // ---- stage gathered x tile into LDS via async global->LDS copies (ASYNCcnt)
  {
#pragma unroll
    for (int u = 0; u < 4; ++u) {
      int unit = tid + u * 256;        // 1024 units of 128 B
      int row  = unit >> 4;
      int seg  = unit & 15;
      const __bf16* src = xb + (size_t)tok_s[row] * DMODEL + seg * 64;
      uint32_t dstl = (uint32_t)(uintptr_t)(&xs[row * DMODEL + seg * 64]);
#pragma unroll
      for (int j = 0; j < 8; ++j) {
        uint64_t ga = (uint64_t)(uintptr_t)(src + j * 8);
        uint32_t la = dstl + j * 16;
        asm volatile("global_load_async_to_lds_b128 %0, %1, off"
                     :: "v"(la), "v"(ga) : "memory");
      }
    }
    asm volatile("s_wait_asynccnt 0x0" ::: "memory");
  }
  __syncthreads();

  int wv    = tid >> 5;
  int lane  = tid & 31;
  int lh    = lane >> 4;               // half-wave select
  int l15   = lane & 15;
  int mg    = wv >> 2;                 // 0..1 -> m-tiles {2mg, 2mg+1}
  int fgr   = wv & 3;                  // 0..3 -> f/d-tile pair index

  int mrow[2];                         // A row index (within 64-token tile)
#pragma unroll
  for (int i = 0; i < 2; ++i) mrow[i] = (2 * mg + i) * 16 + l15;

  // ---- GEMM1: h = relu(x_tile @ W_down[e] + b_down[e]), 2x2 tiles per wave
  int fcol[2];
  const __bf16* bp1[2];
#pragma unroll
  for (int j = 0; j < 2; ++j) {
    fcol[j] = (2 * fgr + j) * 16 + l15;
    bp1[j]  = wd + ((size_t)e * FDIM + fcol[j]) * DMODEL;
  }

  v8f acc[2][2] = {};
  for (int kt = 0; kt < DMODEL / 32; ++kt) {
    int kb = kt * 32;
    v16bf a[2], b[2];
#pragma unroll
    for (int i = 0; i < 2; ++i) {
      const __bf16* ap = &xs[mrow[i] * DMODEL + kb + lh * 16];
      a[i] = *(const v16bf*)ap;
    }
#pragma unroll
    for (int j = 0; j < 2; ++j)
      b[j] = *(const v16bf*)(bp1[j] + kb + lh * 16);
#pragma unroll
    for (int i = 0; i < 2; ++i)
#pragma unroll
      for (int j = 0; j < 2; ++j)
        acc[i][j] = __builtin_amdgcn_wmma_f32_16x16x32_bf16(false, a[i], false, b[j],
                                                            (short)0, acc[i][j], false, false);
  }
#pragma unroll
  for (int j = 0; j < 2; ++j) {
    float bd = b_down[e * FDIM + fcol[j]];
#pragma unroll
    for (int i = 0; i < 2; ++i)
#pragma unroll
      for (int r = 0; r < 8; ++r) {    // C layout: n = l15, m = r + 8*lh
        float v = acc[i][j][r] + bd;
        v = v > 0.f ? v : 0.f;
        hsh[((2 * mg + i) * 16 + r + 8 * lh) * FDIM + fcol[j]] = (__bf16)v;
      }
  }
  __syncthreads();

  // ---- GEMM2: y = h @ W_up[e] + b_up[e]; 8 chunks of 2 d-tiles per wave
  for (int c = 0; c < 8; ++c) {
    int dcol[2];
    const __bf16* bp2[2];
#pragma unroll
    for (int j = 0; j < 2; ++j) {
      dcol[j] = (c * 8 + fgr * 2 + j) * 16 + l15;
      bp2[j]  = wu + ((size_t)e * DMODEL + dcol[j]) * FDIM;
    }
    v8f acc2[2][2] = {};
#pragma unroll
    for (int kt = 0; kt < 4; ++kt) {
      v16bf a2[2], b2[2];
#pragma unroll
      for (int i = 0; i < 2; ++i) {
        const __bf16* hp = &hsh[mrow[i] * FDIM + kt * 32 + lh * 16];
        a2[i] = *(const v16bf*)hp;
      }
#pragma unroll
      for (int j = 0; j < 2; ++j)
        b2[j] = *(const v16bf*)(bp2[j] + kt * 32 + lh * 16);
#pragma unroll
      for (int i = 0; i < 2; ++i)
#pragma unroll
        for (int j = 0; j < 2; ++j)
          acc2[i][j] = __builtin_amdgcn_wmma_f32_16x16x32_bf16(false, a2[i], false, b2[j],
                                                               (short)0, acc2[i][j], false, false);
    }
#pragma unroll
    for (int j = 0; j < 2; ++j) {
      float bu = b_up[e * DMODEL + dcol[j]];
#pragma unroll
      for (int i = 0; i < 2; ++i)
#pragma unroll
        for (int r = 0; r < 8; ++r) {
          int m16  = r + 8 * lh;
          int trow = (2 * mg + i) * 16 + m16;
          float val = SCALE * g_s[trow] * (acc2[i][j][r] + bu);
          atomicAdd(&out[(size_t)tok_s[trow] * DMODEL + dcol[j]], val);
        }
    }
  }
}

// ---------------- launch ----------------

extern "C" void kernel_launch(void* const* d_in, const int* in_sizes, int n_in,
                              void* d_out, int out_size, void* d_ws, size_t ws_size,
                              hipStream_t stream) {
  const float* x       = (const float*)d_in[0];
  const float* noise   = (const float*)d_in[1];
  const float* router  = (const float*)d_in[2];
  const float* w_noise = (const float*)d_in[3];
  const float* W_down  = (const float*)d_in[4];
  const float* b_down  = (const float*)d_in[5];
  const float* W_up    = (const float*)d_in[6];
  const float* b_up    = (const float*)d_in[7];
  float* out = (float*)d_out;

  uint8_t* w = (uint8_t*)d_ws;
  const size_t en4 = (size_t)NEXP * N_TOK * sizeof(int);       // 512 KB
  int*    counts  = (int*)w;                                   // 32 B (pad to 256)
  int*    tok_ids = (int*)(w + 256);
  float*  gate_v  = (float*)(w + 256 + en4);
  __bf16* xb      = (__bf16*)(w + 256 + 2 * en4);              // 256-byte aligned
  __bf16* wd_t    = xb + (size_t)N_TOK * DMODEL;               // 32 MB in
  __bf16* wu_t    = wd_t + (size_t)NEXP * FDIM * DMODEL;       // +2 MB
  (void)in_sizes; (void)n_in; (void)out_size; (void)ws_size;

  k_zero_counts<<<1, 32, 0, stream>>>(counts);
  k_prep_x<<<(N_TOK * DMODEL / 4) / 256, 256, 0, stream>>>(x, xb, out);
  k_prep_wd<<<(NEXP * FDIM * DMODEL) / 256, 256, 0, stream>>>(W_down, wd_t);
  k_prep_wu<<<(NEXP * DMODEL * FDIM) / 256, 256, 0, stream>>>(W_up, wu_t);
  k_gate<<<N_TOK / 8, 256, 0, stream>>>(x, noise, router, w_noise,
                                        counts, tok_ids, gate_v);
  k_expert<<<NEXP * (N_TOK / BTOK), 256, 0, stream>>>(xb, wd_t, wu_t, b_down, b_up,
                                                      counts, tok_ids, gate_v, out);
}
